// PointNetEncoder_1236950581979
// MI455X (gfx1250) — compile-verified, matrix operand order checked by reading, not run on hardware
//
#include <hip/hip_runtime.h>
#include <hip/hip_bf16.h>

typedef __attribute__((ext_vector_type(16))) _Float16 v16h;
typedef __attribute__((ext_vector_type(8)))  float    v8f;

#define BCL 16
#define N1  2048
#define M1  1024
#define N2  1024
#define M2  256
#define KNB 64

// ---------------------------------------------------------------------------
// WMMA fragment loader.
// 16-bit A-matrix 16x32 layout (ISA 7.12.2): lane L -> row m = L&15, half = L>>4;
// VGPR j (holding halves 2j,2j+1): Kbase = (j<4) ? half*8+2j : 16+half*8+2(j-4).
// For B we store weights column-major [N][Kpad] so the same loader applies
// (lane -> output column, VGPRs -> K pairs). Per lane this touches two
// contiguous 16B spans, which the compiler fuses into global_load_b128 pairs.
// ---------------------------------------------------------------------------
__device__ __forceinline__ v16h load_frag(const _Float16* base, int stride) {
  int lane = threadIdx.x & 31;
  int hs = lane >> 4;
  int m  = lane & 15;
  const _Float16* row = base + (size_t)m * stride;
  v16h f;
#pragma unroll
  for (int j = 0; j < 8; ++j) {
    int kb = (j < 4) ? (hs * 8 + 2 * j) : (16 + hs * 8 + 2 * (j - 4));
    f[2 * j]     = row[kb];
    f[2 * j + 1] = row[kb + 1];
  }
  return f;
}

__device__ __forceinline__ v8f wmma_f16(v16h a, v16h b, v8f c) {
  return __builtin_amdgcn_wmma_f32_16x16x32_f16(false, a, false, b, (short)0, c,
                                                false, false);
}

// ---------------------------------------------------------------------------
// Weight prep: f32 row-major [K][N] -> f16 column-major [N][Kp], zero padded.
// ---------------------------------------------------------------------------
__global__ void convert_w_kernel(const float* __restrict__ W, _Float16* __restrict__ out,
                                 int Kdim, int Kp, int Ncols) {
  int i = blockIdx.x * 256 + threadIdx.x;
  if (i >= Ncols * Kp) return;
  int n = i / Kp, k = i % Kp;
  out[i] = (k < Kdim) ? (_Float16)W[(size_t)k * Ncols + n] : (_Float16)0.f;
}

// ---------------------------------------------------------------------------
// Farthest point sampling: one block per cloud, distances in registers,
// block-wide argmax via LDS tree reduction.
// ---------------------------------------------------------------------------
template <int NPTS, int MSEL>
__global__ __launch_bounds__(256) void fps_kernel(const float* __restrict__ pos,
                                                  int* __restrict__ idx_out) {
  constexpr int P = NPTS / 256;
  __shared__ float sval[256];
  __shared__ int   sidx[256];
  __shared__ float lp[3];
  __shared__ int   s_last;
  int tid = threadIdx.x;
  const float* base = pos + (size_t)blockIdx.x * NPTS * 3;
  int* out = idx_out + (size_t)blockIdx.x * MSEL;
  float d2r[P], px[P], py[P], pz[P];
#pragma unroll
  for (int j = 0; j < P; ++j) {
    int p = tid * P + j;
    d2r[j] = 3.4e38f;
    px[j] = base[p * 3 + 0];
    py[j] = base[p * 3 + 1];
    pz[j] = base[p * 3 + 2];
  }
  if (tid == 0) s_last = 0;
  __syncthreads();
  for (int it = 0; it < MSEL; ++it) {
    int last = s_last;
    if (tid == 0) out[it] = last;
    if (tid < 3) lp[tid] = base[last * 3 + tid];
    __syncthreads();
    float lx = lp[0], ly = lp[1], lz = lp[2];
    float bestv = -1.f;
    int   besti = 0;
#pragma unroll
    for (int j = 0; j < P; ++j) {
      float dx = px[j] - lx, dy = py[j] - ly, dz = pz[j] - lz;
      float d = dx * dx + dy * dy + dz * dz;
      d2r[j] = fminf(d2r[j], d);
      if (d2r[j] > bestv) { bestv = d2r[j]; besti = tid * P + j; }
    }
    sval[tid] = bestv;
    sidx[tid] = besti;
    __syncthreads();
    for (int s = 128; s > 0; s >>= 1) {
      if (tid < s && sval[tid + s] > sval[tid]) {
        sval[tid] = sval[tid + s];
        sidx[tid] = sidx[tid + s];
      }
      __syncthreads();
    }
    if (tid == 0) s_last = sidx[0];
    __syncthreads();
  }
}

__global__ void gather_pos_kernel(const float* __restrict__ pos, const int* __restrict__ idx,
                                  float* __restrict__ posq, int M, int Npts, int total) {
  int i = blockIdx.x * 256 + threadIdx.x;
  if (i >= total) return;
  int b = i / M;
  const float* p = pos + ((size_t)b * Npts + idx[i]) * 3;
  posq[(size_t)i * 3 + 0] = p[0];
  posq[(size_t)i * 3 + 1] = p[1];
  posq[(size_t)i * 3 + 2] = p[2];
}

// ---------------------------------------------------------------------------
// Up-to-K nearest neighbors within radius: one block per query. Distances live
// in registers; 64 rounds of block argmin with claim masking. -1 == invalid.
// ---------------------------------------------------------------------------
template <int NPTS>
__global__ __launch_bounds__(256) void knn_kernel(const float* __restrict__ pos,
                                                  const float* __restrict__ posq,
                                                  int* __restrict__ nidx, int M, float r2) {
  constexpr int P = NPTS / 256;
  __shared__ float sval[256];
  __shared__ int   sidx[256];
  __shared__ float pq[3];
  int tid = threadIdx.x;
  int bq = blockIdx.x;
  int b = bq / M;
  const float* base = pos + (size_t)b * NPTS * 3;
  int* out = nidx + (size_t)bq * KNB;
  if (tid < 3) pq[tid] = posq[(size_t)bq * 3 + tid];
  __syncthreads();
  float qx = pq[0], qy = pq[1], qz = pq[2];
  float d2r[P];
#pragma unroll
  for (int j = 0; j < P; ++j) {
    int p = tid * P + j;
    float dx = base[p * 3] - qx, dy = base[p * 3 + 1] - qy, dz = base[p * 3 + 2] - qz;
    float d = dx * dx + dy * dy + dz * dz;
    d2r[j] = (d <= r2) ? d : 3.4e38f;
  }
  for (int k = 0; k < KNB; ++k) {
    float bestv = 3.4e38f;
    int   bestj = -1;
#pragma unroll
    for (int j = 0; j < P; ++j)
      if (d2r[j] < bestv) { bestv = d2r[j]; bestj = j; }
    sval[tid] = bestv;
    sidx[tid] = (bestj >= 0) ? tid * P + bestj : -1;
    __syncthreads();
    for (int s = 128; s > 0; s >>= 1) {
      if (tid < s && sval[tid + s] < sval[tid]) {
        sval[tid] = sval[tid + s];
        sidx[tid] = sidx[tid + s];
      }
      __syncthreads();
    }
    int win = (sval[0] < 3.3e38f) ? sidx[0] : -1;
    if (tid == 0) out[k] = win;
    if (win >= 0 && (win / P) == tid) d2r[win % P] = 3.4e38f;
    __syncthreads();
  }
}

// ---------------------------------------------------------------------------
// WMMA dense layers over a 64-row LDS tile (one query, 64 neighbors).
// ---------------------------------------------------------------------------
template <int KDIM, int COUT>
__device__ __forceinline__ void dense_relu_64(const _Float16* in, int in_stride,
                                              const _Float16* w, const float* bias,
                                              _Float16* out, int out_stride,
                                              int wave, int lane) {
  constexpr int NT = COUT / 16;
  int hs = lane >> 4;
  for (int t = wave; t < 4 * NT; t += 8) {
    int mt = t / NT, nt = t % NT;
    v8f acc = {};
#pragma unroll
    for (int k0 = 0; k0 < KDIM; k0 += 32) {
      if (k0 + 32 < KDIM)
        __builtin_prefetch(w + (size_t)(nt * 16) * KDIM + k0 + 32, 0, 3);
      v16h a  = load_frag(in + (size_t)(mt * 16) * in_stride + k0, in_stride);
      v16h bb = load_frag(w + (size_t)(nt * 16) * KDIM + k0, KDIM);
      acc = wmma_f16(a, bb, acc);
    }
    int n = nt * 16 + (lane & 15);
    float bv = bias[n];
#pragma unroll
    for (int r = 0; r < 8; ++r) {
      int m = mt * 16 + r + 8 * hs;
      float v = acc[r] + bv;
      v = v > 0.f ? v : 0.f;
      out[(size_t)m * out_stride + n] = (_Float16)v;
    }
  }
}

template <int KDIM, int COUT>
__device__ __forceinline__ void dense_maxagg_64(const _Float16* in, int in_stride,
                                                const _Float16* w, const float* bias,
                                                const int* validf, float* x_out,
                                                int wave, int lane) {
  constexpr int NT = COUT / 16;
  int hs = lane >> 4;
  for (int nt = wave; nt < NT; nt += 8) {
    int n = nt * 16 + (lane & 15);
    float bv = bias[n];
    float colmax = -3.4e38f;
#pragma unroll
    for (int mt = 0; mt < 4; ++mt) {
      v8f acc = {};
#pragma unroll
      for (int k0 = 0; k0 < KDIM; k0 += 32) {
        v16h a  = load_frag(in + (size_t)(mt * 16) * in_stride + k0, in_stride);
        v16h bb = load_frag(w + (size_t)(nt * 16) * KDIM + k0, KDIM);
        acc = wmma_f16(a, bb, acc);
      }
#pragma unroll
      for (int r = 0; r < 8; ++r) {
        int m = mt * 16 + r + 8 * hs;
        float v = acc[r] + bv;
        if (validf[m]) colmax = fmaxf(colmax, v);
      }
    }
    float other = __shfl_xor(colmax, 16, 32);
    colmax = fmaxf(colmax, other);
    if (hs == 0) x_out[n] = colmax;
  }
}

// ---------------------------------------------------------------------------
// Fused SAModule body: gather neighbor features -> 2x dense+ReLU -> dense ->
// masked max over the 64 neighbors. One workgroup (8 waves) per query.
// ---------------------------------------------------------------------------
template <bool HAS_X, int CPREV, int IN_PAD, int C1, int C2, int C3>
__global__ __launch_bounds__(256) void sa_fused_kernel(
    const float* __restrict__ pos, const float* __restrict__ posq,
    const int* __restrict__ nidx, const float* __restrict__ x_in,
    const _Float16* __restrict__ w1, const float* __restrict__ b1,
    const _Float16* __restrict__ w2, const float* __restrict__ b2,
    const _Float16* __restrict__ w3, const float* __restrict__ b3,
    float* __restrict__ x_out, int M, int Npts) {
  constexpr int IN_DIM = HAS_X ? CPREV + 3 : 3;
  constexpr int FS = IN_PAD + 8, S1 = C1 + 8, S2 = C2 + 8;
  __shared__ _Float16 feat[64 * FS];
  __shared__ _Float16 h1[64 * S1];
  __shared__ _Float16 h2[64 * S2];
  __shared__ int validf[64];
  __shared__ float pq[3];

  int tid = threadIdx.x, lane = tid & 31, wave = tid >> 5;
  int bq = blockIdx.x;
  int b = bq / M;
  const float* posC = pos + (size_t)b * Npts * 3;
  const int* nb = nidx + (size_t)bq * KNB;
  if (tid < 3) pq[tid] = posq[(size_t)bq * 3 + tid];
  if (tid < 64) validf[tid] = (nb[tid] >= 0) ? 1 : 0;
  __syncthreads();

  // Gather features: [x_j | pos_j - pos_i], zero padded to IN_PAD.
  for (int e = tid; e < 64 * IN_PAD; e += 256) {
    int r = e / IN_PAD, c = e % IN_PAD;
    int n = nb[r];
    float v = 0.f;
    if (n >= 0) {
      if constexpr (HAS_X) {
        const float* xC = x_in + (size_t)b * Npts * CPREV;
        if (c < CPREV)       v = xC[(size_t)n * CPREV + c];
        else if (c < IN_DIM) v = posC[(size_t)n * 3 + (c - CPREV)] - pq[c - CPREV];
      } else {
        if (c < 3) v = posC[(size_t)n * 3 + c] - pq[c];
      }
    }
    feat[r * FS + c] = (_Float16)v;
  }
  __syncthreads();

  dense_relu_64<IN_PAD, C1>(feat, FS, w1, b1, h1, S1, wave, lane);
  __syncthreads();
  dense_relu_64<C1, C2>(h1, S1, w2, b2, h2, S2, wave, lane);
  __syncthreads();
  dense_maxagg_64<C2, C3>(h2, S2, w3, b3, validf, x_out + (size_t)bq * C3, wave, lane);
}

// ---------------------------------------------------------------------------
// Generic WMMA GEMM: out = act(A[R x Kp] @ Wcol[N x Kp]^T + bias).
// A row-major f16, stride Kp. Each wave computes one 16-row x 32-col block
// (two 16x16 WMMA tiles sharing the A fragment): halves A traffic and gives
// two independent WMMAs per K step. Ncols must be a multiple of 32.
// ---------------------------------------------------------------------------
__global__ __launch_bounds__(256) void gemm_kernel(const _Float16* __restrict__ A,
                                                   const _Float16* __restrict__ W,
                                                   const float* __restrict__ bias,
                                                   _Float16* __restrict__ outh,
                                                   float* __restrict__ outf,
                                                   int R, int Kp, int Ncols, int relu) {
  int lane = threadIdx.x & 31, wave = threadIdx.x >> 5;
  int NT2 = Ncols / 32;
  int ntiles = (R / 16) * NT2;
  int t = blockIdx.x * 8 + wave;
  if (t >= ntiles) return;  // wave-uniform
  int mt = t / NT2, np = t % NT2;
  const _Float16* Arow = A + (size_t)(mt * 16) * Kp;
  const _Float16* Wc0  = W + (size_t)(np * 32) * Kp;
  const _Float16* Wc1  = W + (size_t)(np * 32 + 16) * Kp;
  v8f acc0 = {}, acc1 = {};
  for (int k0 = 0; k0 < Kp; k0 += 32) {
    if (k0 + 32 < Kp) {
      __builtin_prefetch(Arow + k0 + 32, 0, 3);
      __builtin_prefetch(Wc0 + k0 + 32, 0, 3);
      __builtin_prefetch(Wc1 + k0 + 32, 0, 3);
    }
    v16h a  = load_frag(Arow + k0, Kp);
    v16h b0 = load_frag(Wc0 + k0, Kp);
    v16h b1 = load_frag(Wc1 + k0, Kp);
    acc0 = wmma_f16(a, b0, acc0);
    acc1 = wmma_f16(a, b1, acc1);
  }
  int n0 = np * 32 + (lane & 15);
  int n1 = n0 + 16;
  int hs = lane >> 4;
  float bv0 = bias ? bias[n0] : 0.f;
  float bv1 = bias ? bias[n1] : 0.f;
#pragma unroll
  for (int r = 0; r < 8; ++r) {
    int m = mt * 16 + r + 8 * hs;
    float v0 = acc0[r] + bv0;
    float v1 = acc1[r] + bv1;
    if (relu) {
      v0 = v0 > 0.f ? v0 : 0.f;
      v1 = v1 > 0.f ? v1 : 0.f;
    }
    if (outh) {
      outh[(size_t)m * Ncols + n0] = (_Float16)v0;
      outh[(size_t)m * Ncols + n1] = (_Float16)v1;
    }
    if (outf) {
      outf[(size_t)m * Ncols + n0] = v0;
      outf[(size_t)m * Ncols + n1] = v1;
    }
  }
}

// concat(x2, pos2) -> f16 rows padded to 288
__global__ void concat3_kernel(const float* __restrict__ x2, const float* __restrict__ pos2,
                               _Float16* __restrict__ cat, int rows) {
  int i = blockIdx.x * 256 + threadIdx.x;
  if (i >= rows * 288) return;
  int r = i / 288, c = i % 288;
  float v = 0.f;
  if (c < 256)      v = x2[(size_t)r * 256 + c];
  else if (c < 259) v = pos2[(size_t)r * 3 + (c - 256)];
  cat[i] = (_Float16)v;
}

__global__ void maxpool_kernel(const _Float16* __restrict__ h, _Float16* __restrict__ g,
                               int rowsPerCloud, int C, int total) {
  int i = blockIdx.x * 256 + threadIdx.x;
  if (i >= total) return;
  int b = i / C, c = i % C;
  const _Float16* base = h + (size_t)b * rowsPerCloud * C + c;
  float m = -3.4e38f;
  for (int r = 0; r < rowsPerCloud; ++r) m = fmaxf(m, (float)base[(size_t)r * C]);
  g[i] = (_Float16)m;
}

// ---------------------------------------------------------------------------
extern "C" void kernel_launch(void* const* d_in, const int* in_sizes, int n_in,
                              void* d_out, int out_size, void* d_ws, size_t ws_size,
                              hipStream_t stream) {
  (void)in_sizes; (void)n_in; (void)out_size; (void)ws_size;
  const float* pos = (const float*)d_in[0];
  // params1: W/b x3 at 1..6, params2 at 7..12, params3 at 13..18, head at 19..24
  const float* W1a = (const float*)d_in[1];  const float* B1a = (const float*)d_in[2];
  const float* W1b = (const float*)d_in[3];  const float* B1b = (const float*)d_in[4];
  const float* W1c = (const float*)d_in[5];  const float* B1c = (const float*)d_in[6];
  const float* W2a = (const float*)d_in[7];  const float* B2a = (const float*)d_in[8];
  const float* W2b = (const float*)d_in[9];  const float* B2b = (const float*)d_in[10];
  const float* W2c = (const float*)d_in[11]; const float* B2c = (const float*)d_in[12];
  const float* W3a = (const float*)d_in[13]; const float* B3a = (const float*)d_in[14];
  const float* W3b = (const float*)d_in[15]; const float* B3b = (const float*)d_in[16];
  const float* W3c = (const float*)d_in[17]; const float* B3c = (const float*)d_in[18];
  const float* W4a = (const float*)d_in[19]; const float* B4a = (const float*)d_in[20];
  const float* W4b = (const float*)d_in[21]; const float* B4b = (const float*)d_in[22];
  const float* W4c = (const float*)d_in[23]; const float* B4c = (const float*)d_in[24];

  // bump allocator over d_ws
  size_t off = 0;
  auto alloc = [&](size_t bytes) -> void* {
    off = (off + 255) & ~(size_t)255;
    void* p = (char*)d_ws + off;
    off += bytes;
    return p;
  };

  _Float16* w1a = (_Float16*)alloc(32  * 64   * 2);
  _Float16* w1b = (_Float16*)alloc(64  * 64   * 2);
  _Float16* w1c = (_Float16*)alloc(64  * 128  * 2);
  _Float16* w2a = (_Float16*)alloc(160 * 128  * 2);
  _Float16* w2b = (_Float16*)alloc(128 * 128  * 2);
  _Float16* w2c = (_Float16*)alloc(128 * 256  * 2);
  _Float16* w3a = (_Float16*)alloc(288 * 256  * 2);
  _Float16* w3b = (_Float16*)alloc(256 * 512  * 2);
  _Float16* w3c = (_Float16*)alloc(512 * 1024 * 2);
  _Float16* w4a = (_Float16*)alloc(1024 * 512 * 2);
  _Float16* w4b = (_Float16*)alloc(512 * 256  * 2);
  _Float16* w4c = (_Float16*)alloc(256 * 64   * 2);

  int*      idx1  = (int*)alloc((size_t)BCL * M1 * 4);
  float*    pos1  = (float*)alloc((size_t)BCL * M1 * 3 * 4);
  int*      nidx1 = (int*)alloc((size_t)BCL * M1 * KNB * 4);
  float*    x1    = (float*)alloc((size_t)BCL * M1 * 128 * 4);
  int*      idx2  = (int*)alloc((size_t)BCL * M2 * 4);
  float*    pos2  = (float*)alloc((size_t)BCL * M2 * 3 * 4);
  int*      nidx2 = (int*)alloc((size_t)BCL * M2 * KNB * 4);
  float*    x2    = (float*)alloc((size_t)BCL * M2 * 256 * 4);
  _Float16* cat3  = (_Float16*)alloc((size_t)BCL * M2 * 288 * 2);
  _Float16* h3a   = (_Float16*)alloc((size_t)BCL * M2 * 256 * 2);
  _Float16* h3b   = (_Float16*)alloc((size_t)BCL * M2 * 512 * 2);
  _Float16* h3c   = (_Float16*)alloc((size_t)BCL * M2 * 1024 * 2);
  _Float16* g     = (_Float16*)alloc((size_t)BCL * 1024 * 2);
  _Float16* h4a   = (_Float16*)alloc((size_t)BCL * 512 * 2);
  _Float16* h4b   = (_Float16*)alloc((size_t)BCL * 256 * 2);

  auto cw = [&](const float* W, _Float16* o, int K, int Kp, int Nc) {
    int tot = Nc * Kp;
    convert_w_kernel<<<(tot + 255) / 256, 256, 0, stream>>>(W, o, K, Kp, Nc);
  };
  cw(W1a, w1a, 3,    32,   64);
  cw(W1b, w1b, 64,   64,   64);
  cw(W1c, w1c, 64,   64,   128);
  cw(W2a, w2a, 131,  160,  128);
  cw(W2b, w2b, 128,  128,  128);
  cw(W2c, w2c, 128,  128,  256);
  cw(W3a, w3a, 259,  288,  256);
  cw(W3b, w3b, 256,  256,  512);
  cw(W3c, w3c, 512,  512,  1024);
  cw(W4a, w4a, 1024, 1024, 512);
  cw(W4b, w4b, 512,  512,  256);
  cw(W4c, w4c, 256,  256,  64);

  // ---- SA module 1 --------------------------------------------------------
  fps_kernel<N1, M1><<<BCL, 256, 0, stream>>>(pos, idx1);
  gather_pos_kernel<<<(BCL * M1 + 255) / 256, 256, 0, stream>>>(pos, idx1, pos1, M1, N1,
                                                               BCL * M1);
  knn_kernel<N1><<<BCL * M1, 256, 0, stream>>>(pos, pos1, nidx1, M1, 0.04f);
  sa_fused_kernel<false, 0, 32, 64, 64, 128><<<BCL * M1, 256, 0, stream>>>(
      pos, pos1, nidx1, nullptr, w1a, B1a, w1b, B1b, w1c, B1c, x1, M1, N1);

  // ---- SA module 2 --------------------------------------------------------
  fps_kernel<N2, M2><<<BCL, 256, 0, stream>>>(pos1, idx2);
  gather_pos_kernel<<<(BCL * M2 + 255) / 256, 256, 0, stream>>>(pos1, idx2, pos2, M2, N2,
                                                               BCL * M2);
  knn_kernel<N2><<<BCL * M2, 256, 0, stream>>>(pos1, pos2, nidx2, M2, 0.16f);
  sa_fused_kernel<true, 128, 160, 128, 128, 256><<<BCL * M2, 256, 0, stream>>>(
      pos1, pos2, nidx2, x1, w2a, B2a, w2b, B2b, w2c, B2c, x2, M2, N2);

  // ---- MLP3 + global max pool --------------------------------------------
  int rows3 = BCL * M2;  // 4096
  concat3_kernel<<<(rows3 * 288 + 255) / 256, 256, 0, stream>>>(x2, pos2, cat3, rows3);
  {
    int nt = (rows3 / 16) * (256 / 32);
    gemm_kernel<<<(nt + 7) / 8, 256, 0, stream>>>(cat3, w3a, B3a, h3a, nullptr,
                                                  rows3, 288, 256, 1);
  }
  {
    int nt = (rows3 / 16) * (512 / 32);
    gemm_kernel<<<(nt + 7) / 8, 256, 0, stream>>>(h3a, w3b, B3b, h3b, nullptr,
                                                  rows3, 256, 512, 1);
  }
  {
    int nt = (rows3 / 16) * (1024 / 32);
    gemm_kernel<<<(nt + 7) / 8, 256, 0, stream>>>(h3b, w3c, B3c, h3c, nullptr,
                                                  rows3, 512, 1024, 0);
  }
  maxpool_kernel<<<(BCL * 1024 + 255) / 256, 256, 0, stream>>>(h3c, g, M2, 1024,
                                                               BCL * 1024);

  // ---- Head MLP -----------------------------------------------------------
  {
    int nt = (BCL / 16) * (512 / 32);
    gemm_kernel<<<(nt + 7) / 8, 256, 0, stream>>>(g, w4a, B4a, h4a, nullptr,
                                                  BCL, 1024, 512, 1);
  }
  {
    int nt = (BCL / 16) * (256 / 32);
    gemm_kernel<<<(nt + 7) / 8, 256, 0, stream>>>(h4a, w4b, B4b, h4b, nullptr,
                                                  BCL, 512, 256, 1);
  }
  {
    int nt = (BCL / 16) * (64 / 32);
    gemm_kernel<<<(nt + 7) / 8, 256, 0, stream>>>(h4b, w4c, B4c, nullptr, (float*)d_out,
                                                  BCL, 256, 64, 0);
  }
}
